// VSSBlock_45234595561772
// MI455X (gfx1250) — compile-verified, matrix operand order checked by reading, not run on hardware
//
#include <hip/hip_runtime.h>
#include <hip/hip_bf16.h>

typedef __attribute__((ext_vector_type(16))) _Float16 v16h;
typedef __attribute__((ext_vector_type(8)))  float    v8f;

namespace {
constexpr int Tt = 4, Bb = 4, Hh = 56, Ww = 56, Cc = 96;
constexpr int DI = 192;          // d_inner
constexpr int Ns = 16;           // d_state
constexpr int Rr = 6;            // dt_rank
constexpr int Kk = 4;            // scan directions
constexpr int Ll = Hh * Ww;      // 3136
constexpr int TB = Tt * Bb;      // 16
constexpr int Mtok = TB * Ll;    // 50176  (divisible by 128)
constexpr int XP = 48;           // padded x_dbl width (R + 2N = 38 -> 48)
constexpr int FC = 4 * Cc;       // 384
}

#define DEV static __device__ __forceinline__

// ---------------- WMMA fragment loaders (16x16x32 f16, wave32) ----------------
// A-frag (16x32, row-major source): lane L -> row m=L&15, half=L>>4.
// element i<8  : k = k0 + half*8 + i
// element i>=8 : k = k0 + 16 + half*8 + (i-8)
// B-frag is loaded from B^T rows (weight rows) with the identical mapping.
template<bool MASKK>
DEV v16h load_frag(const _Float16* __restrict__ rowbase, int k0, int half, int K) {
  v16h f;
  if (!MASKK) {
    const _Float16* p = rowbase + k0 + half * 8;
#pragma unroll
    for (int i = 0; i < 8; ++i) f[i] = p[i];
#pragma unroll
    for (int i = 0; i < 8; ++i) f[i + 8] = p[16 + i];
  } else {
#pragma unroll
    for (int i = 0; i < 8; ++i) {
      int k = k0 + half * 8 + i;
      f[i] = (k < K) ? rowbase[k] : (_Float16)0.f;
    }
#pragma unroll
    for (int i = 0; i < 8; ++i) {
      int k = k0 + 16 + half * 8 + i;
      f[i + 8] = (k < K) ? rowbase[k] : (_Float16)0.f;
    }
  }
  return f;
}

// Wave computes a 16 x (NT*16) tile of  C[M,N] = A[M,K] * Bt[N,K]^T
template<int NT, bool MASKK>
DEV void gemm_core(const _Float16* __restrict__ A, int lda,
                   const _Float16* __restrict__ Bt, int ldb,
                   int K, int row0, int col0, int nvalid, v8f* acc) {
  const int lane = threadIdx.x & 31;
  const int mr = lane & 15, half = lane >> 4;
  const _Float16* arow = A + (size_t)(row0 + mr) * lda;
  for (int k0 = 0; k0 < K; k0 += 32) {
    v16h af = load_frag<MASKK>(arow, k0, half, K);
#pragma unroll
    for (int t = 0; t < NT; ++t) {
      int bn = col0 + t * 16 + mr;
      v16h bf;
      if (bn < nvalid) {
        bf = load_frag<MASKK>(Bt + (size_t)bn * ldb, k0, half, K);
      } else {
        v16h z = {};
        bf = z;
      }
      acc[t] = __builtin_amdgcn_wmma_f32_16x16x32_f16(
          false, af, false, bf, (short)0, acc[t], false, false);
    }
  }
}

// D layout: VGPR r -> row = row0 + r + 8*(lane>>4), col = col0 + t*16 + (lane&15)

DEV float silu(float v)  { return v / (1.f + __expf(-v)); }
DEV float softp(float v) { return v > 20.f ? v : log1pf(__expf(v)); }
DEV float gelu(float v)  { return 0.5f * v * (1.f + erff(v * 0.70710678118654752f)); }

// ---------------- f32 -> f16 convert ----------------
__global__ void k_cvt(const float* __restrict__ s, _Float16* __restrict__ d, int n) {
  int i = blockIdx.x * blockDim.x + threadIdx.x;
  if (i < n) d[i] = (_Float16)s[i];
}

// ---------------- in_proj: (M,96)x(384,96)^T ; split xi / silu(z) ----------------
__global__ __launch_bounds__(256) void k_inproj(const _Float16* __restrict__ x16,
                                                const _Float16* __restrict__ w16,
                                                _Float16* __restrict__ xi16,
                                                _Float16* __restrict__ zs16) {
  constexpr int NT = 4;
  v8f acc[NT] = {};
  int wv = threadIdx.x >> 5;
  int row0 = blockIdx.y * 128 + wv * 16;
  int col0 = blockIdx.x * (NT * 16);
  gemm_core<NT, false>(x16, Cc, w16, Cc, Cc, row0, col0, 1 << 30, acc);
  int lane = threadIdx.x & 31, nc = lane & 15, hf = lane >> 4;
#pragma unroll
  for (int t = 0; t < NT; ++t)
#pragma unroll
    for (int r = 0; r < 8; ++r) {
      int row = row0 + r + 8 * hf, col = col0 + t * 16 + nc;
      float v = acc[t][r];
      if (col < DI) xi16[(size_t)row * DI + col] = (_Float16)v;
      else          zs16[(size_t)row * DI + (col - DI)] = (_Float16)silu(v);
    }
}

// ---------------- depthwise 3x3 conv + bias + SiLU; emit (h,w) and (w,h) orders ----
__global__ void k_conv(const _Float16* __restrict__ xi16,
                       const float* __restrict__ cw, const float* __restrict__ cb,
                       _Float16* __restrict__ xc0, _Float16* __restrict__ xc1) {
  int gid = blockIdx.x * blockDim.x + threadIdx.x;
  if (gid >= Mtok * DI) return;
  int d = gid % DI;
  int rest = gid / DI;
  int w = rest % Ww; rest /= Ww;
  int h = rest % Hh; int tb = rest / Hh;
  const _Float16* base = xi16 + (size_t)tb * Ll * DI;
  float a = 0.f;
#pragma unroll
  for (int i = 0; i < 3; ++i) {
    int hh = h + i - 1;
    if ((unsigned)hh >= (unsigned)Hh) continue;
#pragma unroll
    for (int j = 0; j < 3; ++j) {
      int ww = w + j - 1;
      if ((unsigned)ww >= (unsigned)Ww) continue;
      a += (float)base[((size_t)hh * Ww + ww) * DI + d] * cw[d * 9 + i * 3 + j];
    }
  }
  a += cb[d];
  _Float16 hv = (_Float16)silu(a);
  xc0[((size_t)tb * Ll + h * Ww + w) * DI + d] = hv;   // row-major scan order
  xc1[((size_t)tb * Ll + w * Hh + h) * DI + d] = hv;   // transposed scan order
}

// ---------------- x_proj (per direction): (M,192)x(38,192)^T -> x_dbl (padded 48) ----
__global__ __launch_bounds__(256) void k_xproj(const _Float16* __restrict__ xc,
                                               const _Float16* __restrict__ wk,
                                               _Float16* __restrict__ xdbl_k) {
  constexpr int NT = 3;
  v8f acc[NT] = {};
  int wv = threadIdx.x >> 5;
  int row0 = blockIdx.y * 128 + wv * 16;
  int col0 = blockIdx.x * (NT * 16);
  gemm_core<NT, false>(xc, DI, wk, DI, DI, row0, col0, Rr + 2 * Ns, acc);
  int lane = threadIdx.x & 31, nc = lane & 15, hf = lane >> 4;
#pragma unroll
  for (int t = 0; t < NT; ++t)
#pragma unroll
    for (int r = 0; r < 8; ++r) {
      int row = row0 + r + 8 * hf, col = col0 + t * 16 + nc;
      if (col < Rr + 2 * Ns) xdbl_k[(size_t)row * XP + col] = (_Float16)acc[t][r];
    }
}

// ---------------- dt_proj (per direction): (M,6)x(192,6)^T + bias -> softplus ------
__global__ __launch_bounds__(256) void k_dtproj(const _Float16* __restrict__ xdbl_k,
                                                const _Float16* __restrict__ dtw_k,
                                                const float* __restrict__ dtb_k,
                                                _Float16* __restrict__ delta_k) {
  constexpr int NT = 4;
  v8f acc[NT] = {};
  int wv = threadIdx.x >> 5;
  int row0 = blockIdx.y * 128 + wv * 16;
  int col0 = blockIdx.x * (NT * 16);
  gemm_core<NT, true>(xdbl_k, XP, dtw_k, Rr, Rr, row0, col0, 1 << 30, acc);
  int lane = threadIdx.x & 31, nc = lane & 15, hf = lane >> 4;
#pragma unroll
  for (int t = 0; t < NT; ++t)
#pragma unroll
    for (int r = 0; r < 8; ++r) {
      int row = row0 + r + 8 * hf, col = col0 + t * 16 + nc;
      delta_k[(size_t)row * DI + col] = (_Float16)softp(acc[t][r] + dtb_k[col]);
    }
}

// ================= Tensor Data Mover support (B/C tile staging) =================
#if defined(__HIP_DEVICE_COMPILE__) && defined(__gfx1250__) && \
    __has_builtin(__builtin_amdgcn_tensor_load_to_lds) &&       \
    __has_builtin(__builtin_amdgcn_s_wait_tensorcnt)
#define SCAN_TDM 1
typedef __attribute__((ext_vector_type(4))) unsigned int u32x4;
typedef __attribute__((ext_vector_type(8))) int i32x8;
typedef __attribute__((ext_vector_type(4))) int i32x4;

// 2-D f16 tile load: rows x cols elements, row stride = stride_elems, into LDS.
// D# group0: count=1 | lds_addr | global_addr[56:0] | type=2 ("image")
// D# group1: data_size=1 (2B) | tensor_dim0=cols | tensor_dim1=rows |
//            tile_dim0=cols | tile_dim1=rows | tensor_dim0_stride=stride_elems
DEV void tdm_load_tile_f16(const _Float16* gsrc, unsigned lds_off,
                           int rows, int cols, int stride_elems) {
  unsigned long long ga = (unsigned long long)(uintptr_t)gsrc;
  u32x4 g0;
  g0[0] = 1u;                                                 // count=1, user desc
  g0[1] = lds_off;                                            // LDS byte address
  g0[2] = (unsigned)ga;                                       // global_addr[31:0]
  g0[3] = (unsigned)((ga >> 32) & 0x01ffffffu) | (2u << 30);  // [56:32] | type=2
  i32x8 g1;
  g1[0] = 1 << 16;                                            // data_size=2B; mask=0
  g1[1] = (cols & 0xffff) << 16;                              // tensor_dim0 lo16
  g1[2] = (int)(((unsigned)cols >> 16) & 0xffffu) | ((rows & 0xffff) << 16);
  g1[3] = (int)(((unsigned)rows >> 16) & 0xffffu) | ((cols & 0xffff) << 16);
  g1[4] = rows & 0xffff;                                      // tile_dim1 (dim2=0)
  g1[5] = stride_elems;                                       // dim0_stride[31:0]
  g1[6] = 0;
  g1[7] = 0;
  i32x4 z4 = {0, 0, 0, 0};
#if __has_include(<hip/amd_detail/amd_gfx1250_TDM.h>)
  i32x8 z8 = {0, 0, 0, 0, 0, 0, 0, 0};
  __builtin_amdgcn_tensor_load_to_lds(g0, g1, z4, z4, z8, 0);   // therock 6-arg form
#else
  __builtin_amdgcn_tensor_load_to_lds(g0, g1, z4, z4, 0);       // ROCm 7.2 5-arg form
#endif
}
#endif

// ---------------- selective scan: one lane per (tb,k,d) ----------------
// B/C tile per chunk (CH steps x 32 f16, stride XP) staged to LDS via TDM,
// double-buffered, DMA for chunk c+1 overlapped with compute of chunk c.
__global__ __launch_bounds__(192) void k_scan(const _Float16* __restrict__ xc0,
                                              const _Float16* __restrict__ xc1,
                                              const _Float16* __restrict__ xdbl,
                                              const _Float16* __restrict__ delta,
                                              const float* __restrict__ A_logs,
                                              const float* __restrict__ Ds,
                                              _Float16* __restrict__ ys) {
  constexpr int CH = 8;                       // steps per chunk (3136 % 8 == 0)
  constexpr int NC = Ll / CH;                 // 392 chunks
  const int tb = blockIdx.x;
  const int k  = blockIdx.y;
  const int d  = threadIdx.x;                 // 0..191
  const bool rev = (k >= 2);

  const _Float16* u_base  = ((k & 1) ? xc1 : xc0) + (size_t)tb * Ll * DI + d;
  const _Float16* dl_base = delta + ((size_t)k * Mtok + (size_t)tb * Ll) * DI + d;
  const _Float16* bc_base = xdbl  + ((size_t)k * Mtok + (size_t)tb * Ll) * XP;
  _Float16* y_base = ys + ((size_t)k * Mtok + (size_t)tb * Ll) * DI + d;

  // first (lowest-memory) row of chunk c, ascending rows; +Rr skips the dts cols
  auto chunk_ptr = [&](int c) -> const _Float16* {
    int ldfirst = rev ? (Ll - (c + 1) * CH) : (c * CH);
    return bc_base + (size_t)ldfirst * XP + Rr;
  };

  float Arow[Ns];
#pragma unroll
  for (int n = 0; n < Ns; ++n)
    Arow[n] = -__expf(A_logs[(size_t)(k * DI + d) * Ns + n]);
  const float Dv = Ds[k * DI + d];

  float h[Ns];
#pragma unroll
  for (int n = 0; n < Ns; ++n) h[n] = 0.f;

  __shared__ _Float16 sBC[2][CH][2 * Ns];
#if defined(SCAN_TDM)
  const bool issuer = (threadIdx.x >> 5) == 0;   // wave 0 drives the TDM
  unsigned ldsaddr0 = (unsigned)(uintptr_t)&sBC[0][0][0];
  unsigned ldsaddr1 = (unsigned)(uintptr_t)&sBC[1][0][0];
  if (issuer) tdm_load_tile_f16(chunk_ptr(0), ldsaddr0, CH, 2 * Ns, XP);
#endif

  for (int c = 0; c < NC; ++c) {
#if defined(SCAN_TDM)
    if (issuer) {
      if (c + 1 < NC) {
        tdm_load_tile_f16(chunk_ptr(c + 1), ((c + 1) & 1) ? ldsaddr1 : ldsaddr0,
                          CH, 2 * Ns, XP);
        __builtin_amdgcn_s_wait_tensorcnt(1);    // chunk c done, c+1 in flight
      } else {
        __builtin_amdgcn_s_wait_tensorcnt(0);
      }
    }
#else
    {
      const _Float16* gp = chunk_ptr(c);
      for (int i = threadIdx.x; i < CH * 2 * Ns; i += blockDim.x) {
        int r = i >> 5, j = i & 31;
        sBC[c & 1][r][j] = gp[(size_t)r * XP + j];
      }
    }
#endif
    if (c + 1 < NC) {   // prefetch next chunk's u/delta stream
      int lpf = rev ? (Ll - 1 - (c + 1) * CH) : ((c + 1) * CH);
      __builtin_prefetch(u_base + (size_t)lpf * DI, 0, 1);
      __builtin_prefetch(dl_base + (size_t)lpf * DI, 0, 1);
    }
    __syncthreads();
    const _Float16(*buf)[2 * Ns] = sBC[c & 1];
#pragma unroll
    for (int s = 0; s < CH; ++s) {
      int ls = c * CH + s;
      int ld = rev ? (Ll - 1 - ls) : ls;        // data index of this step
      int srow = rev ? (CH - 1 - s) : s;        // LDS row (tiles loaded ascending)
      float u  = (float)u_base[(size_t)ld * DI];
      float dt = (float)dl_base[(size_t)ld * DI];
      float du = dt * u;
      float y = 0.f;
#pragma unroll
      for (int n = 0; n < Ns; ++n) {
        h[n] = h[n] * __expf(dt * Arow[n]) + du * (float)buf[srow][n];
        y += h[n] * (float)buf[srow][Ns + n];
      }
      y_base[(size_t)ld * DI] = (_Float16)(y + Dv * u);   // store un-reversed
    }
    __syncthreads();
  }
}

// ---------------- cross-merge + LayerNorm + SiLU(z) gate ----------------
__global__ __launch_bounds__(192) void k_merge(const _Float16* __restrict__ ys,
                                               const _Float16* __restrict__ zs,
                                               const float* __restrict__ g,
                                               const float* __restrict__ bta,
                                               _Float16* __restrict__ ya) {
  const int m = blockIdx.x;            // token, row-major (h,w)
  const int d = threadIdx.x;           // 0..191
  const int tb = m / Ll, l = m % Ll;
  const int hh = l / Ww, wv2 = l % Ww;
  const int l2 = wv2 * Hh + hh;        // position in (w,h)-ordered buffers
  float y = (float)ys[((size_t)(0 * TB + tb) * Ll + l ) * DI + d]
          + (float)ys[((size_t)(2 * TB + tb) * Ll + l ) * DI + d]
          + (float)ys[((size_t)(1 * TB + tb) * Ll + l2) * DI + d]
          + (float)ys[((size_t)(3 * TB + tb) * Ll + l2) * DI + d];

  float s = y, s2 = y * y;
#pragma unroll
  for (int o = 16; o > 0; o >>= 1) {
    s  += __shfl_down(s,  o, 32);
    s2 += __shfl_down(s2, o, 32);
  }
  __shared__ float ps[6], ps2[6], mb[2];
  int wvx = threadIdx.x >> 5, lane = threadIdx.x & 31;
  if (lane == 0) { ps[wvx] = s; ps2[wvx] = s2; }
  __syncthreads();
  if (threadIdx.x == 0) {
    float a = 0.f, b2 = 0.f;
    for (int i = 0; i < 6; ++i) { a += ps[i]; b2 += ps2[i]; }
    float mu = a / DI, var = b2 / DI - mu * mu;
    mb[0] = mu; mb[1] = rsqrtf(var + 1e-5f);
  }
  __syncthreads();
  float yn = (y - mb[0]) * mb[1] * g[d] + bta[d];
  float o  = yn * (float)zs[(size_t)m * DI + d];
  ya[(size_t)m * DI + d] = (_Float16)o;
}

// ---------------- out_proj: (M,192)x(96,192)^T + residual x ----------------
__global__ __launch_bounds__(256) void k_outproj(const _Float16* __restrict__ ya,
                                                 const _Float16* __restrict__ w16,
                                                 const float* __restrict__ x,
                                                 float* __restrict__ xf32,
                                                 _Float16* __restrict__ xf16) {
  constexpr int NT = 3;
  v8f acc[NT] = {};
  int wv = threadIdx.x >> 5;
  int row0 = blockIdx.y * 128 + wv * 16;
  int col0 = blockIdx.x * (NT * 16);
  gemm_core<NT, false>(ya, DI, w16, DI, DI, row0, col0, 1 << 30, acc);
  int lane = threadIdx.x & 31, nc = lane & 15, hf = lane >> 4;
#pragma unroll
  for (int t = 0; t < NT; ++t)
#pragma unroll
    for (int r = 0; r < 8; ++r) {
      int row = row0 + r + 8 * hf, col = col0 + t * 16 + nc;
      float v = acc[t][r] + x[(size_t)row * Cc + col];
      xf32[(size_t)row * Cc + col] = v;
      xf16[(size_t)row * Cc + col] = (_Float16)v;
    }
}

// ---------------- fc1: (M,96)x(384,96)^T + bias -> GELU ----------------
__global__ __launch_bounds__(256) void k_fc1(const _Float16* __restrict__ xf16,
                                             const _Float16* __restrict__ w16,
                                             const float* __restrict__ b,
                                             _Float16* __restrict__ h1) {
  constexpr int NT = 4;
  v8f acc[NT] = {};
  int wv = threadIdx.x >> 5;
  int row0 = blockIdx.y * 128 + wv * 16;
  int col0 = blockIdx.x * (NT * 16);
  gemm_core<NT, false>(xf16, Cc, w16, Cc, Cc, row0, col0, 1 << 30, acc);
  int lane = threadIdx.x & 31, nc = lane & 15, hf = lane >> 4;
#pragma unroll
  for (int t = 0; t < NT; ++t)
#pragma unroll
    for (int r = 0; r < 8; ++r) {
      int row = row0 + r + 8 * hf, col = col0 + t * 16 + nc;
      h1[(size_t)row * FC + col] = (_Float16)gelu(acc[t][r] + b[col]);
    }
}

// ---------------- fc2: (M,384)x(96,384)^T + bias + residual -> d_out ----------------
__global__ __launch_bounds__(256) void k_fc2(const _Float16* __restrict__ h1,
                                             const _Float16* __restrict__ w16,
                                             const float* __restrict__ b,
                                             const float* __restrict__ xf32,
                                             float* __restrict__ out) {
  constexpr int NT = 3;
  v8f acc[NT] = {};
  int wv = threadIdx.x >> 5;
  int row0 = blockIdx.y * 128 + wv * 16;
  int col0 = blockIdx.x * (NT * 16);
  gemm_core<NT, false>(h1, FC, w16, FC, FC, row0, col0, 1 << 30, acc);
  int lane = threadIdx.x & 31, nc = lane & 15, hf = lane >> 4;
#pragma unroll
  for (int t = 0; t < NT; ++t)
#pragma unroll
    for (int r = 0; r < 8; ++r) {
      int row = row0 + r + 8 * hf, col = col0 + t * 16 + nc;
      out[(size_t)row * Cc + col] = acc[t][r] + b[col] + xf32[(size_t)row * Cc + col];
    }
}

// =====================================================================
extern "C" void kernel_launch(void* const* d_in, const int* in_sizes, int n_in,
                              void* d_out, int out_size, void* d_ws, size_t ws_size,
                              hipStream_t stream) {
  const float* x       = (const float*)d_in[0];
  const float* in_w    = (const float*)d_in[1];
  const float* conv_w  = (const float*)d_in[2];
  const float* conv_b  = (const float*)d_in[3];
  const float* xproj_w = (const float*)d_in[4];
  const float* dtw     = (const float*)d_in[5];
  const float* dtb     = (const float*)d_in[6];
  const float* A_logs  = (const float*)d_in[7];
  const float* Ds      = (const float*)d_in[8];
  const float* ln_g    = (const float*)d_in[9];
  const float* ln_b    = (const float*)d_in[10];
  const float* outp_w  = (const float*)d_in[11];
  const float* fc1_w   = (const float*)d_in[12];
  const float* fc1_b   = (const float*)d_in[13];
  const float* fc2_w   = (const float*)d_in[14];
  const float* fc2_b   = (const float*)d_in[15];
  float* out = (float*)d_out;

  char* ws = (char*)d_ws;
  size_t off = 0;
  auto alloc = [&](size_t bytes) -> char* {
    off = (off + 255) & ~(size_t)255;
    char* p = ws + off;
    off += bytes;
    return p;
  };
  const size_t HS = sizeof(_Float16);
  _Float16* w16_in   = (_Float16*)alloc((size_t)2 * DI * Cc * HS);
  _Float16* w16_xp   = (_Float16*)alloc((size_t)Kk * (Rr + 2 * Ns) * DI * HS);
  _Float16* w16_dtw  = (_Float16*)alloc((size_t)Kk * DI * Rr * HS);
  _Float16* w16_op   = (_Float16*)alloc((size_t)Cc * DI * HS);
  _Float16* w16_fc1  = (_Float16*)alloc((size_t)FC * Cc * HS);
  _Float16* w16_fc2  = (_Float16*)alloc((size_t)Cc * FC * HS);
  _Float16* x16      = (_Float16*)alloc((size_t)Mtok * Cc * HS);
  _Float16* xi16     = (_Float16*)alloc((size_t)Mtok * DI * HS);
  _Float16* zs16     = (_Float16*)alloc((size_t)Mtok * DI * HS);
  _Float16* xc0      = (_Float16*)alloc((size_t)Mtok * DI * HS);
  _Float16* xc1      = (_Float16*)alloc((size_t)Mtok * DI * HS);
  _Float16* xdbl16   = (_Float16*)alloc((size_t)Kk * Mtok * XP * HS);
  _Float16* delta16  = (_Float16*)alloc((size_t)Kk * Mtok * DI * HS);
  _Float16* ys16     = (_Float16*)alloc((size_t)Kk * Mtok * DI * HS);
  _Float16* ya16     = (_Float16*)alloc((size_t)Mtok * DI * HS);
  float*    xf32     = (float*)   alloc((size_t)Mtok * Cc * sizeof(float));
  _Float16* xf16     = (_Float16*)alloc((size_t)Mtok * Cc * HS);
  _Float16* h116     = (_Float16*)alloc((size_t)Mtok * FC * HS);

  auto cvt = [&](const float* s, _Float16* d, int n) {
    k_cvt<<<(n + 255) / 256, 256, 0, stream>>>(s, d, n);
  };
  cvt(in_w,    w16_in,  2 * DI * Cc);
  cvt(xproj_w, w16_xp,  Kk * (Rr + 2 * Ns) * DI);
  cvt(dtw,     w16_dtw, Kk * DI * Rr);
  cvt(outp_w,  w16_op,  Cc * DI);
  cvt(fc1_w,   w16_fc1, FC * Cc);
  cvt(fc2_w,   w16_fc2, Cc * FC);
  cvt(x,       x16,     Mtok * Cc);

  const int MB = Mtok / 128;  // 392 row-blocks

  // in_proj -> xi, silu(z)
  k_inproj<<<dim3(2 * DI / 64, MB), 256, 0, stream>>>(x16, w16_in, xi16, zs16);

  // depthwise conv + silu, both scan orders
  k_conv<<<(Mtok * DI + 255) / 256, 256, 0, stream>>>(xi16, conv_w, conv_b, xc0, xc1);

  // x_proj and dt_proj per direction
  for (int k = 0; k < Kk; ++k) {
    const _Float16* xc = (k & 1) ? xc1 : xc0;
    k_xproj<<<dim3(1, MB), 256, 0, stream>>>(
        xc, w16_xp + (size_t)k * (Rr + 2 * Ns) * DI, xdbl16 + (size_t)k * Mtok * XP);
    k_dtproj<<<dim3(DI / 64, MB), 256, 0, stream>>>(
        xdbl16 + (size_t)k * Mtok * XP, w16_dtw + (size_t)k * DI * Rr,
        dtb + (size_t)k * DI, delta16 + (size_t)k * Mtok * DI);
  }

  // selective scan (4 directions x 16 batch, 192 channels per block, TDM-staged B/C)
  k_scan<<<dim3(TB, Kk), DI, 0, stream>>>(xc0, xc1, xdbl16, delta16, A_logs, Ds, ys16);

  // cross-merge + LayerNorm + gate
  k_merge<<<Mtok, DI, 0, stream>>>(ys16, zs16, ln_g, ln_b, ya16);

  // out_proj + residual 1
  k_outproj<<<dim3(Cc / 48, MB), 256, 0, stream>>>(ya16, w16_op, x, xf32, xf16);

  // MLP
  k_fc1<<<dim3(FC / 64, MB), 256, 0, stream>>>(xf16, w16_fc1, fc1_b, h116);
  k_fc2<<<dim3(Cc / 48, MB), 256, 0, stream>>>(h116, w16_fc2, fc2_b, xf32, out);
}
